// RNN_AE_18863496364500
// MI455X (gfx1250) — compile-verified, matrix operand order checked by reading, not run on hardware
//
#include <hip/hip_runtime.h>

// ---------------------------------------------------------------------------
// Persistent wave32 WMMA RNN-autoencoder for MI455X (gfx1250).
//   - 64 persistent workgroups x 256 threads (8 waves), 64 batch rows each
//   - activations [x|h] live in LDS as bf16 (K padded 204 -> 224)
//   - weight column-strips live in registers as v16bf WMMA B-fragments
//   - encoder x_t tiles DMA'd into LDS by the Tensor Data Mover
//     (tensor_load_to_lds, TENSORcnt), pipelined one step ahead
//   - per-step batch-wide BN stats + categorical argmax masks accumulate via
//     L2 atomics into per-step scratch slots; one global atomic-counter
//     barrier per timestep (monotonic, per-instance counters -> no reset race)
// ---------------------------------------------------------------------------

typedef __bf16 v16bf __attribute__((ext_vector_type(16)));
typedef float  v8f   __attribute__((ext_vector_type(8)));
typedef unsigned int v4u __attribute__((ext_vector_type(4)));
typedef int          v4i __attribute__((ext_vector_type(4)));
typedef int          v8i __attribute__((ext_vector_type(8)));

#define B_    4096
#define T_    96
#define F_    76
#define H_    128
#define KPAD  224      // 7 * 32 (K padded; cols 204..223 are zero)
#define KVAL  204      // F_ + H_
#define ROWS  64       // batch rows per workgroup
#define NWG   64       // 64 * 64 = 4096 rows
#define NTHR  256      // 8 waves of 32
#define EPS_  1e-5f

#if defined(__has_builtin)
#if __has_builtin(__builtin_amdgcn_tensor_load_to_lds)
#define HAVE_TDM 1
#endif
#endif
#ifndef HAVE_TDM
#define HAVE_TDM 0
#endif

__constant__ int catb_[10] = {1, 7, 11, 16, 19, 26, 32, 37, 41, 47};
__constant__ int cate_[10] = {6, 10, 15, 18, 25, 31, 36, 40, 46, 50};

__device__ __forceinline__ __bf16 f2bf(float f) {
  unsigned u = __builtin_bit_cast(unsigned, f);
  u += 0x7fffu + ((u >> 16) & 1u);                 // round-to-nearest-even
  unsigned short s = (unsigned short)(u >> 16);
  return __builtin_bit_cast(__bf16, s);
}
__device__ __forceinline__ float bf2f(__bf16 b) {
  unsigned short s = __builtin_bit_cast(unsigned short, b);
  unsigned u = ((unsigned)s) << 16;
  return __builtin_bit_cast(float, u);
}

__device__ __forceinline__ v8f zero8() {
  v8f z = {0.f, 0.f, 0.f, 0.f, 0.f, 0.f, 0.f, 0.f};
  return z;
}
__device__ __forceinline__ v16bf zero16() {
  v16bf z;
#pragma unroll
  for (int j = 0; j < 16; ++j) z[j] = f2bf(0.f);
  return z;
}

__device__ __forceinline__ v8f wmma_bf16(v16bf a, v16bf b, v8f c) {
  // D(f32 16x16) = A(bf16 16x32) * B(bf16 32x16) + C   -> v_wmma_f32_16x16x32_bf16
  return __builtin_amdgcn_wmma_f32_16x16x32_bf16(
      /*neg_a=*/false, a, /*neg_b=*/false, b,
      /*c_mod=*/(short)0, c, /*reuse_a=*/false, /*reuse_b=*/false);
}

// A fragment from LDS: lane<16 -> row r0+lane, K {0..7,16..23}; lane>=16 -> K {8..15,24..31}
__device__ __forceinline__ v16bf load_afrag(const __bf16* rowp, int k0, int hf) {
  int koff = k0 + hf * 8;
  v16bf a;
#pragma unroll
  for (int j = 0; j < 8; ++j) {
    a[j]     = rowp[koff + j];
    a[8 + j] = rowp[koff + 16 + j];
  }
  return a;
}

// B fragment from global f32 weight W[K][ncols] (mirrored layout: lane owns column)
__device__ __forceinline__ v16bf load_bfrag(const float* __restrict__ W, int ncols,
                                            int k0, int n0, int lane) {
  int hf   = lane >> 4;
  int col  = n0 + (lane & 15);
  int koff = k0 + hf * 8;
  v16bf b;
#pragma unroll
  for (int j = 0; j < 8; ++j) {
    int k1 = koff + j;
    int k2 = koff + 16 + j;
    float f1 = (k1 < KVAL && col < ncols) ? W[k1 * ncols + col] : 0.f;
    float f2 = (k2 < KVAL && col < ncols) ? W[k2 * ncols + col] : 0.f;
    b[j]     = f2bf(f1);
    b[8 + j] = f2bf(f2);
  }
  return b;
}

#if HAVE_TDM
// Issue a TDM load of the 64x76 f32 x_t tile (row stride T_*F_) into LDS.
// D# per CDNA5 ISA ch.8: group0 = {count/type, lds_addr, global_addr},
// group1 = {data_size, tensor dims, tile dims, dim0 stride}. 2D tensor ->
// groups 2/3 zero. One wave per WG issues; completion via TENSORcnt.
__device__ __forceinline__ void tdm_issue_x(const float* __restrict__ x,
                                            int rowBase, int t, unsigned lds_off) {
  unsigned long long ga =
      (unsigned long long)(const void*)(x + ((unsigned long long)rowBase * T_ + t) * F_);
  v4u g0;
  g0[0] = 1u;                                   // count=1 (valid user descriptor)
  g0[1] = lds_off;                              // lds_addr (bytes)
  g0[2] = (unsigned)(ga & 0xffffffffu);         // global_addr[31:0]
  g0[3] = (unsigned)((ga >> 32) & 0x01ffffffu)  // global_addr[56:32]
          | (2u << 30);                         // type = 2 ("image")
  v8i g1;
  g1[0] = 0x20000;                              // data_size = 2 (4 bytes), no flags
  g1[1] = (int)(76u << 16);                     // tensor_dim0 = 76   [63:48]
  g1[2] = (int)(64u << 16);                     // tensor_dim1 = 64   [95:80]
  g1[3] = (int)(76u << 16);                     // tile_dim0 = 76     [127:112]
  g1[4] = 64;                                   // tile_dim1 = 64     [143:128]
  g1[5] = T_ * F_;                              // tensor_dim0_stride [191:160]
  g1[6] = 0;
  g1[7] = 0;
  v4i z4 = {0, 0, 0, 0};
#if defined(__clang_major__) && (__clang_major__ >= 23)
  v8i z8 = {0, 0, 0, 0, 0, 0, 0, 0};
  __builtin_amdgcn_tensor_load_to_lds(g0, g1, z4, z4, z8, 0);
#else
  __builtin_amdgcn_tensor_load_to_lds(g0, g1, z4, z4, 0);
#endif
}
#endif

// Device-wide barrier: one monotonic counter per barrier instance (zeroed by init kernel)
__device__ __forceinline__ void gbar(unsigned* __restrict__ ctr) {
  __syncthreads();
  if (threadIdx.x == 0) {
    __threadfence();
    atomicAdd(ctr, 1u);
    while (atomicAdd(ctr, 0u) < (unsigned)NWG) __builtin_amdgcn_s_sleep(1);
    __threadfence();
  }
  __syncthreads();
}

__global__ void ws_zero_kernel(unsigned* __restrict__ p, int n) {
  int i = blockIdx.x * blockDim.x + threadIdx.x;
  if (i < n) p[i] = 0u;
}

__global__ __launch_bounds__(NTHR, 1) void rnn_ae_kernel(
    const float* __restrict__ x,
    const float* __restrict__ W_enc, const float* __restrict__ b_enc,
    const float* __restrict__ g_enc, const float* __restrict__ be_enc,
    const float* __restrict__ W_dec, const float* __restrict__ b_dec,
    const float* __restrict__ g_dec, const float* __restrict__ be_dec,
    const float* __restrict__ W_out, const float* __restrict__ b_out,
    float* __restrict__ out,
    float* __restrict__ stats,    // [2*T][256] : sums(128) | sumsq(128) per step
    unsigned* __restrict__ gmask, // [T][16]    : onehot column masks per group
    unsigned* __restrict__ barctr) {

  __shared__ __bf16  actA[ROWS][KPAD];  // [x|h] (enc) / [din|h or new_h] (dec), bf16
  __shared__ float   xstage[ROWS][F_];  // raw f32 x_t tile (TDM destination)
  __shared__ __bf16  obuf[ROWS][80];    // decoder output tile (padded 76->80)
  __shared__ float   sstat[256];        // staged BN stats
  __shared__ unsigned smask[16];        // per-group argmax column masks

  const int tid  = threadIdx.x;
  const int lane = tid & 31;
  const int wv   = tid >> 5;    // wave id 0..7 (wave-uniform)
  const int hf   = lane >> 4;   // half-wave select
  const int l15  = lane & 15;
  const int rowBase = blockIdx.x * ROWS;

  // zero the whole activation buffer once (also establishes zero K-padding & h0=0)
  for (int idx = tid; idx < ROWS * KPAD; idx += NTHR)
    actA[idx / KPAD][idx % KPAD] = f2bf(0.f);
  __syncthreads();

  const int   colH = wv * 16 + l15;       // hidden column owned by this lane
  const float bE  = b_enc[colH];
  const float gE  = g_enc[colH];
  const float beE = be_enc[colH];

  // ---- resident W_enc fragments for this wave's 16-column strip ----
  v16bf benc[7];
#pragma unroll
  for (int k = 0; k < 7; ++k) benc[k] = load_bfrag(W_enc, H_, k * 32, wv * 16, lane);

#if HAVE_TDM
  const unsigned xstage_lds = (unsigned)(unsigned long long)(const void*)&xstage[0][0];
  if (wv == 0) tdm_issue_x(x, rowBase, 0, xstage_lds);   // prime the pipeline
#endif

  // ============================ encoder ============================
  for (int t = 0; t < T_; ++t) {
#if HAVE_TDM
    __builtin_amdgcn_s_wait_tensorcnt(0);   // x_t tile landed in LDS (wave 0)
    __syncthreads();                        // publish to all waves
    for (int idx = tid; idx < ROWS * F_; idx += NTHR) {
      int r = idx / F_, f = idx % F_;
      actA[r][f] = f2bf(xstage[r][f]);      // f32 -> bf16 into GEMM operand
    }
    __syncthreads();                        // xstage consumed; actA ready
    if (wv == 0 && t + 1 < T_)
      tdm_issue_x(x, rowBase, t + 1, xstage_lds);  // overlap DMA with compute
#else
    for (int idx = tid; idx < ROWS * F_; idx += NTHR) {
      int r = idx / F_, f = idx % F_;
      size_t gi = ((size_t)(rowBase + r) * T_ + t) * F_ + f;
      actA[r][f] = f2bf(x[gi]);
      if (t + 1 < T_) __builtin_prefetch(&x[gi + F_], 0, 0);
    }
    __syncthreads();
#endif

    // new_h = tanh([x|h] @ W_enc + b): 4 row tiles x this wave's col tile
    v8f acc[4];
#pragma unroll
    for (int rt = 0; rt < 4; ++rt) acc[rt] = zero8();
#pragma unroll
    for (int k = 0; k < 7; ++k) {
      v16bf bw = benc[k];
#pragma unroll
      for (int rt = 0; rt < 4; ++rt) {
        v16bf af = load_afrag(&actA[rt * 16 + l15][0], k * 32, hf);
        acc[rt] = wmma_bf16(af, bw, acc[rt]);
      }
    }

    // v = tanh(.) + h_old ; partial batch sums for BN
    float cs = 0.f, cq = 0.f;
#pragma unroll
    for (int rt = 0; rt < 4; ++rt)
#pragma unroll
      for (int i = 0; i < 8; ++i) {
        int r = rt * 16 + i + hf * 8;      // C/D layout: M = i (+8 for upper half)
        float nh   = tanhf(acc[rt][i] + bE);
        float hold = bf2f(actA[r][F_ + colH]);
        float v = nh + hold;
        acc[rt][i] = v;
        cs += v; cq += v * v;
      }
    cs += __shfl_xor(cs, 16);
    cq += __shfl_xor(cq, 16);
    if (hf == 0) {
      atomicAdd(&stats[t * 256 + colH], cs);
      atomicAdd(&stats[t * 256 + 128 + colH], cq);
    }
    gbar(&barctr[t]);  // all partials in L2

    sstat[tid] = atomicAdd(&stats[t * 256 + tid], 0.f);  // coherent L2 read
    __syncthreads();
    float m  = sstat[colH] * (1.f / B_);
    float vr = sstat[128 + colH] * (1.f / B_) - m * m;
    float sc = rsqrtf(vr + EPS_) * gE;
#pragma unroll
    for (int rt = 0; rt < 4; ++rt)
#pragma unroll
      for (int i = 0; i < 8; ++i) {
        int r = rt * 16 + i + hf * 8;
        actA[r][F_ + colH] = f2bf((acc[rt][i] - m) * sc + beE);  // h <- BN(v)
      }
    __syncthreads();
  }

  // ============================ decoder ============================
  const float bD  = b_dec[colH];
  const float gD  = g_dec[colH];
  const float beD = be_dec[colH];
  v16bf bdec[7];
#pragma unroll
  for (int k = 0; k < 7; ++k) bdec[k] = load_bfrag(W_dec, H_, k * 32, wv * 16, lane);

  const int colO = wv * 16 + l15;   // output column (waves 0..4 cover 80 padded cols)
  v16bf bout[7];
  float bO = 0.f;
  if (wv < 5) {
#pragma unroll
    for (int k = 0; k < 7; ++k) bout[k] = load_bfrag(W_out, F_, k * 32, wv * 16, lane);
    if (colO < F_) bO = b_out[colO];
  } else {
#pragma unroll
    for (int k = 0; k < 7; ++k) bout[k] = zero16();
  }

  // din0 = 0
  for (int idx = tid; idx < ROWS * F_; idx += NTHR)
    actA[idx / F_][idx % F_] = f2bf(0.f);
  __syncthreads();

  for (int t = 0; t < T_; ++t) {
    // new_h = tanh([din|h] @ W_dec + b)
    v8f acc[4];
#pragma unroll
    for (int rt = 0; rt < 4; ++rt) acc[rt] = zero8();
#pragma unroll
    for (int k = 0; k < 7; ++k) {
      v16bf bw = bdec[k];
#pragma unroll
      for (int rt = 0; rt < 4; ++rt) {
        v16bf af = load_afrag(&actA[rt * 16 + l15][0], k * 32, hf);
        acc[rt] = wmma_bf16(af, bw, acc[rt]);
      }
    }

    v8f nhv[4];
    float cs = 0.f, cq = 0.f;
#pragma unroll
    for (int rt = 0; rt < 4; ++rt)
#pragma unroll
      for (int i = 0; i < 8; ++i) {
        int r = rt * 16 + i + hf * 8;
        float nh   = tanhf(acc[rt][i] + bD);
        float hold = bf2f(actA[r][F_ + colH]);
        float v = nh + hold;
        acc[rt][i] = v;       // keep v for BN after the global barrier
        nhv[rt][i] = nh;
        cs += v; cq += v * v;
      }
    cs += __shfl_xor(cs, 16);
    cq += __shfl_xor(cq, 16);
    if (hf == 0) {
      atomicAdd(&stats[(T_ + t) * 256 + colH], cs);
      atomicAdd(&stats[(T_ + t) * 256 + 128 + colH], cq);
    }
    __syncthreads();  // all reads of actA (GEMM + h_old) complete

    // overwrite h region with new_h (pre-BN) -> input for the output GEMM
#pragma unroll
    for (int rt = 0; rt < 4; ++rt)
#pragma unroll
      for (int i = 0; i < 8; ++i) {
        int r = rt * 16 + i + hf * 8;
        actA[r][F_ + colH] = f2bf(nhv[rt][i]);
      }
    if (tid < 16) smask[tid] = 0u;
    __syncthreads();

    // o = [din|new_h] @ W_out + b_out  (waves 0..4)
    if (wv < 5) {
      v8f ao[4];
#pragma unroll
      for (int rt = 0; rt < 4; ++rt) ao[rt] = zero8();
#pragma unroll
      for (int k = 0; k < 7; ++k) {
        v16bf bw = bout[k];
#pragma unroll
        for (int rt = 0; rt < 4; ++rt) {
          v16bf af = load_afrag(&actA[rt * 16 + l15][0], k * 32, hf);
          ao[rt] = wmma_bf16(af, bw, ao[rt]);
        }
      }
#pragma unroll
      for (int rt = 0; rt < 4; ++rt)
#pragma unroll
        for (int i = 0; i < 8; ++i) {
          int r = rt * 16 + i + hf * 8;
          float o = ao[rt][i] + bO;
          obuf[r][colO] = f2bf(o);
          if (colO < F_)
            out[((size_t)(rowBase + r) * T_ + t) * F_ + colO] = o;
        }
    }
    __syncthreads();  // obuf ready, smask zeroed

    // per-row argmax per categorical group -> OR column bits (WG-local, then global)
    if (tid < ROWS) {
      int r = tid;
#pragma unroll
      for (int g = 0; g < 10; ++g) {
        int b0 = catb_[g], e0 = cate_[g];
        float best = bf2f(obuf[r][b0]);
        int bi = 0;
        for (int f = b0 + 1; f < e0; ++f) {
          float v = bf2f(obuf[r][f]);
          if (v > best) { best = v; bi = f - b0; }
        }
        atomicOr(&smask[g], 1u << bi);
      }
    }
    __syncthreads();
    if (tid < 10) atomicOr(&gmask[t * 16 + tid], smask[tid]);

    gbar(&barctr[T_ + t]);  // BN partials + masks global

    sstat[tid] = atomicAdd(&stats[(T_ + t) * 256 + tid], 0.f);
    if (tid < 10) smask[tid] = atomicOr(&gmask[t * 16 + tid], 0u);
    __syncthreads();

    // h <- BN(new_h + h_old) using global stats
    float m  = sstat[colH] * (1.f / B_);
    float vr = sstat[128 + colH] * (1.f / B_) - m * m;
    float sc = rsqrtf(vr + EPS_) * gD;
#pragma unroll
    for (int rt = 0; rt < 4; ++rt)
#pragma unroll
      for (int i = 0; i < 8; ++i) {
        int r = rt * 16 + i + hf * 8;
        actA[r][F_ + colH] = f2bf((acc[rt][i] - m) * sc + beD);
      }

    // din <- discretized o (global onehot columns for categorical groups)
    for (int idx = tid; idx < ROWS * F_; idx += NTHR) {
      int r = idx / F_, f = idx % F_;
      int g = -1, b0 = 0;
#pragma unroll
      for (int gg = 0; gg < 10; ++gg)
        if (f >= catb_[gg] && f < cate_[gg]) { g = gg; b0 = catb_[gg]; }
      float val = (g >= 0) ? (((smask[g] >> (f - b0)) & 1u) ? 1.f : 0.f)
                           : bf2f(obuf[r][f]);
      actA[r][f] = f2bf(val);
    }
    __syncthreads();
  }
}

extern "C" void kernel_launch(void* const* d_in, const int* in_sizes, int n_in,
                              void* d_out, int out_size, void* d_ws, size_t ws_size,
                              hipStream_t stream) {
  const float* x      = (const float*)d_in[0];
  const float* W_enc  = (const float*)d_in[1];
  const float* b_enc  = (const float*)d_in[2];
  const float* g_enc  = (const float*)d_in[3];
  const float* be_enc = (const float*)d_in[4];
  const float* W_dec  = (const float*)d_in[5];
  const float* b_dec  = (const float*)d_in[6];
  const float* g_dec  = (const float*)d_in[7];
  const float* be_dec = (const float*)d_in[8];
  const float* W_out  = (const float*)d_in[9];
  const float* b_out  = (const float*)d_in[10];
  float* out = (float*)d_out;

  const int STATS_WORDS = 2 * T_ * 256;  // per-step BN sums/sumsq
  const int GMASK_WORDS = T_ * 16;       // per-step onehot masks
  const int CTR_WORDS   = 256;           // 192 barrier instances, rounded up

  float*    stats  = (float*)d_ws;
  unsigned* gmask  = (unsigned*)d_ws + STATS_WORDS;
  unsigned* barctr = (unsigned*)d_ws + STATS_WORDS + GMASK_WORDS;

  const int zwords = STATS_WORDS + GMASK_WORDS + CTR_WORDS;
  ws_zero_kernel<<<(zwords + 255) / 256, 256, 0, stream>>>((unsigned*)d_ws, zwords);
  rnn_ae_kernel<<<NWG, NTHR, 0, stream>>>(x, W_enc, b_enc, g_enc, be_enc,
                                          W_dec, b_dec, g_dec, be_dec, W_out, b_out,
                                          out, stats, gmask, barctr);
  (void)in_sizes; (void)n_in; (void)out_size; (void)ws_size;
}
